// MultiheadSelfAttention_16758962389509
// MI455X (gfx1250) — matrix operand order repaired and checked by reading, not compiled, not run on hardware
//
#include <hip/hip_runtime.h>
#include <hip/hip_bf16.h>

// ---------------------------------------------------------------------------
// MHSA on gfx1250: all matmuls on V_WMMA_F32_16X16X32_F16 (f16 in, f32 acc).
// Compute-bound (~137 GFLOP vs ~80MB traffic). Attention stages K/V tiles
// into LDS with global_load_async_to_lds_b128 (double-buffered, ASYNCcnt).
// ---------------------------------------------------------------------------

typedef __attribute__((ext_vector_type(16))) _Float16 v16h;
typedef __attribute__((ext_vector_type(8)))  _Float16 v8h;
typedef __attribute__((ext_vector_type(8)))  float    v8f;

#define B_  4
#define S_  2048
#define E_  1024
#define H_  16
#define D_  64

static __device__ __forceinline__ v8f wmma_f16(v16h a, v16h b, v8f c) {
  return __builtin_amdgcn_wmma_f32_16x16x32_f16(false, a, false, b,
                                                (short)0, c, false, false);
}

// A-operand fragment (16x32, M=row, K contiguous in memory).
static __device__ __forceinline__ v16h load_a(const _Float16* __restrict__ base,
                                              int stride, int k0) {
  const int lane = threadIdx.x & 31;
  const int m = lane & 15, sel = lane >> 4;
  const _Float16* p = base + (size_t)m * stride + k0 + sel * 8;
  v8h lo = *(const v8h*)p;
  v8h hi = *(const v8h*)(p + 16);
  v16h a;
#pragma unroll
  for (int i = 0; i < 8; ++i) { a[i] = lo[i]; a[8 + i] = hi[i]; }
  return a;
}

// B-operand fragment (32x16). `base` stores B transposed (row = column n of
// B, K contiguous). Lane l: n=l&15, sel=l>>4, two 16-byte loads at K=k0+sel*8 and k0+16+sel*8.
static __device__ __forceinline__ v16h load_b(const _Float16* __restrict__ base,
                                              int stride, int k0) {
  const int lane = threadIdx.x & 31;
  const int n = lane & 15, sel = lane >> 4;
  const _Float16* p = base + (size_t)n * stride + k0 + sel * 8;
  v8h lo = *(const v8h*)p;
  v8h hi = *(const v8h*)(p + 16);
  v16h b;
#pragma unroll
  for (int i = 0; i < 8; ++i) { b[i] = lo[i]; b[8 + i] = hi[i]; }
  return b;
}

static __device__ __forceinline__ float redmax16(float x) {
  x = fmaxf(x, __shfl_xor(x, 1, 16));
  x = fmaxf(x, __shfl_xor(x, 2, 16));
  x = fmaxf(x, __shfl_xor(x, 4, 16));
  x = fmaxf(x, __shfl_xor(x, 8, 16));
  return x;
}
static __device__ __forceinline__ float redsum16(float x) {
  x += __shfl_xor(x, 1, 16);
  x += __shfl_xor(x, 2, 16);
  x += __shfl_xor(x, 4, 16);
  x += __shfl_xor(x, 8, 16);
  return x;
}

// LDS byte offset of a generic pointer into a __shared__ object.
static __device__ __forceinline__ unsigned lds_off(const void* p) {
  return (unsigned)(unsigned long long)
      (const __attribute__((address_space(3))) char*)p;
}

// Issue one 128-byte LDS row fill as 8 async b128 loads (ASYNCcnt-tracked).
static __device__ __forceinline__ void async_row_128B(const _Float16* g,
                                                      unsigned lds) {
#pragma unroll
  for (int i = 0; i < 8; ++i) {
    asm volatile("global_load_async_to_lds_b128 %0, %1, off"
                 :: "v"(lds + i * 16), "v"(g + i * 8) : "memory");
  }
}

static __device__ __forceinline__ void wait_async_group(bool more_inflight) {
  if (more_inflight)
    asm volatile("s_wait_asynccnt 0x8" ::: "memory");  // older group done
  else
    asm volatile("s_wait_asynccnt 0x0" ::: "memory");
}

// ------------------------- conversion kernels ------------------------------

__global__ void k_cvt_f16(const float* __restrict__ in, _Float16* __restrict__ out,
                          int n) {
  int i = blockIdx.x * blockDim.x + threadIdx.x;
  if (i < n) out[i] = (_Float16)in[i];
}

// [H,E,D] f32 -> [H,D,E] f16 (B-operand layout: K=E contiguous)
__global__ void k_tr_w(const float* __restrict__ in, _Float16* __restrict__ out) {
  int i = blockIdx.x * blockDim.x + threadIdx.x;   // total H*E*D = 1M
  int h = i >> 16;
  int rem = i & 65535;
  int d = rem >> 10;
  int e = rem & 1023;
  out[i] = (_Float16)in[(size_t)h * (E_ * D_) + (size_t)e * D_ + d];
}

// [HD,E] f32 -> [E,HD] f16
__global__ void k_tr_wo(const float* __restrict__ in, _Float16* __restrict__ out) {
  int i = blockIdx.x * blockDim.x + threadIdx.x;   // total 1M
  int e = i >> 10;
  int k = i & 1023;
  out[i] = (_Float16)in[(size_t)k * (H_ * D_) + e];
}

// ------------------------- QKV projection ----------------------------------
// grid = (B*H*(S/64), 3): blockIdx.y selects Q/K/V so each wave carries only
// 4 accumulators (no spills) and can double-buffer its fragments.
__global__ void k_qkv(const _Float16* __restrict__ Xh,
                      const _Float16* __restrict__ Wqt,
                      const _Float16* __restrict__ Wkt,
                      const _Float16* __restrict__ Wvt,
                      const float* __restrict__ bq, const float* __restrict__ bk,
                      const float* __restrict__ bv,
                      _Float16* __restrict__ Qh, _Float16* __restrict__ Kh,
                      _Float16* __restrict__ Vt) {
  const int which = blockIdx.y;                       // 0=Q 1=K 2=V
  const _Float16* __restrict__ Wt =
      (which == 0) ? Wqt : (which == 1) ? Wkt : Wvt;
  const float* __restrict__ bias =
      (which == 0) ? bq : (which == 1) ? bk : bv;

  const int lane = threadIdx.x & 31;
  const int wave = threadIdx.x >> 5;
  const int sblk = blockIdx.x & 31;
  const int bh   = blockIdx.x >> 5;
  const int b = bh >> 4, h = bh & 15;
  const int s0 = sblk * 64 + wave * 16;
  const int n = lane & 15, g = lane >> 4;

  const _Float16* xrow = Xh + ((size_t)b * S_ + s0) * E_;

  v8f c[4] = {};
  v16h a = load_a(xrow, E_, 0);
  v16h bf[4];
#pragma unroll
  for (int dj = 0; dj < 4; ++dj)
    bf[dj] = load_b(Wt + (size_t)(h * D_ + dj * 16) * E_, E_, 0);

  for (int k0 = 0; k0 < E_; k0 += 32) {
    const int kn = k0 + 32;
    v16h a2;
    v16h b2[4];
    if (kn < E_) {                       // prefetch next k-step fragments
      a2 = load_a(xrow, E_, kn);
#pragma unroll
      for (int dj = 0; dj < 4; ++dj)
        b2[dj] = load_b(Wt + (size_t)(h * D_ + dj * 16) * E_, E_, kn);
    }
#pragma unroll
    for (int dj = 0; dj < 4; ++dj) c[dj] = wmma_f16(a, bf[dj], c[dj]);
    if (kn < E_) {
      a = a2;
#pragma unroll
      for (int dj = 0; dj < 4; ++dj) bf[dj] = b2[dj];
    }
  }

  const size_t bhS = (size_t)(b * H_ + h) * S_;
  if (which == 2) {                      // V stored transposed [B,H,D,S]
#pragma unroll
    for (int dj = 0; dj < 4; ++dj) {
      const float bv_ = bias[h * D_ + dj * 16 + n];
#pragma unroll
      for (int r = 0; r < 8; ++r) {
        const int m = r + 8 * g;
        const size_t vi =
            ((size_t)(b * H_ + h) * D_ + dj * 16 + n) * (size_t)S_ + s0 + m;
        Vt[vi] = (_Float16)(c[dj][r] + bv_);
      }
    }
  } else {
    _Float16* __restrict__ dst = (which == 0) ? Qh : Kh;
#pragma unroll
    for (int dj = 0; dj < 4; ++dj) {
      const float bv_ = bias[h * D_ + dj * 16 + n];
#pragma unroll
      for (int r = 0; r < 8; ++r) {
        const int m = r + 8 * g;
        dst[(bhS + s0 + m) * D_ + dj * 16 + n] = (_Float16)(c[dj][r] + bv_);
      }
    }
  }
}

// ------------------------- flash attention ---------------------------------
// grid = B*H*(S/64) blocks, 128 threads. K/V tiles for the 64-key block are
// staged cooperatively into LDS with async loads, double-buffered so the next
// block streams in while the current one is computed.
__global__ void k_attn(const _Float16* __restrict__ Qh,
                       const _Float16* __restrict__ Kh,
                       const _Float16* __restrict__ Vt,
                       _Float16* __restrict__ Zc) {
  // [ K 2x(64x64) | V 2x(64x64) | P 4x(16x64) ] halves
  __shared__ _Float16 smem[8192 + 8192 + 4096];
  _Float16* const Kbuf = smem;             // + parity*4096
  _Float16* const Vbuf = smem + 8192;      // + parity*4096
  _Float16* const Pbuf = smem + 16384;     // + wave*1024

  const int tid  = threadIdx.x;
  const int lane = tid & 31;
  const int wave = tid >> 5;
  const int sblk = blockIdx.x & 31;
  const int bh   = blockIdx.x >> 5;
  const int b = bh >> 4, h = bh & 15;
  const int s0 = sblk * 64 + wave * 16;
  const int n = lane & 15, g = lane >> 4;

  const size_t bhS = (size_t)bh * S_;
  const unsigned ldsK = lds_off(Kbuf);
  const unsigned ldsV = lds_off(Vbuf);

  // Each thread stages one 128B row per tile-buffer fill:
  //   waves 0-1 -> K rows (tid = row), waves 2-3 -> V rows (tid-64 = row).
  const int krow = tid;          // valid when wave<2
  const int vrow = tid - 64;     // valid when wave>=2
  const _Float16* const Krow0 = Kh + (bhS + krow) * D_;
  const _Float16* const Vrow0 = Vt + ((size_t)bh * D_ + vrow) * (size_t)S_;

  // ---- stage first K/V block into buffer 0 ----
  if (wave < 2)  async_row_128B(Krow0, ldsK + (unsigned)krow * 128u);
  else           async_row_128B(Vrow0, ldsV + (unsigned)vrow * 128u);

  _Float16* Pl = Pbuf + wave * 1024;
  const _Float16* qrow = Qh + (bhS + s0) * D_;
  const v16h qa0 = load_a(qrow, D_, 0);
  const v16h qa1 = load_a(qrow, D_, 32);

  v8f z[4] = {};
  float mrun[8], lrun[8];
#pragma unroll
  for (int r = 0; r < 8; ++r) { mrun[r] = -1.0e30f; lrun[r] = 0.0f; }

  const float scale = 0.125f;   // 1/sqrt(64)

  for (int t0 = 0; t0 < S_; t0 += 64) {
    const int parity = (t0 >> 6) & 1;
    const int tn = t0 + 64;
    const bool have_next = (tn < S_);

    // ---- kick off async staging of the NEXT key block ----
    if (have_next) {
      const unsigned po = (unsigned)(parity ^ 1) * 8192u;  // bytes
      if (wave < 2)
        async_row_128B(Krow0 + (size_t)tn * D_, ldsK + po + (unsigned)krow * 128u);
      else
        async_row_128B(Vrow0 + tn, ldsV + po + (unsigned)vrow * 128u);
    }
    wait_async_group(have_next);   // current block's tiles are in LDS
    __syncthreads();

    const _Float16* Kt = Kbuf + parity * 4096;   // [t][d] 64x64
    const _Float16* Vq = Vbuf + parity * 4096;   // [d][t] 64x64

    // ---- scores: 4 tiles of 16 keys, K-dim = D (two 32-chunks) ----
    v8f s[4];
#pragma unroll
    for (int j = 0; j < 4; ++j) {
      v8f cc = {};
      cc = wmma_f16(qa0, load_b(Kt + (16 * j) * D_, D_, 0), cc);
      cc = wmma_f16(qa1, load_b(Kt + (16 * j) * D_, D_, 32), cc);
      s[j] = cc * scale;
    }

    // ---- online softmax update ----
    float mnew[8], alpha[8], rs[8];
#pragma unroll
    for (int r = 0; r < 8; ++r) {
      float v = fmaxf(fmaxf(s[0][r], s[1][r]), fmaxf(s[2][r], s[3][r]));
      v = redmax16(v);
      mnew[r]  = fmaxf(mrun[r], v);
      alpha[r] = __expf(mrun[r] - mnew[r]);
      rs[r] = 0.0f;
    }
#pragma unroll
    for (int j = 0; j < 4; ++j)
#pragma unroll
      for (int r = 0; r < 8; ++r) {
        float p = __expf(s[j][r] - mnew[r]);
        s[j][r] = p;
        rs[r] += p;
      }
#pragma unroll
    for (int r = 0; r < 8; ++r) {
      rs[r]   = redsum16(rs[r]);
      lrun[r] = lrun[r] * alpha[r] + rs[r];
      mrun[r] = mnew[r];
    }
#pragma unroll
    for (int dj = 0; dj < 4; ++dj)
#pragma unroll
      for (int r = 0; r < 8; ++r) z[dj][r] *= alpha[r];

    // ---- reshape P (C-frag) -> A-frag via LDS ----
#pragma unroll
    for (int j = 0; j < 4; ++j)
#pragma unroll
      for (int r = 0; r < 8; ++r)
        Pl[(r + 8 * g) * 64 + 16 * j + n] = (_Float16)s[j][r];
    __syncthreads();
    const v16h pa0 = load_a(Pl, 64, 0);
    const v16h pa1 = load_a(Pl, 64, 32);

    // ---- Z += P @ V (V tile in LDS, [d][t]: K=t contiguous) ----
#pragma unroll
    for (int dj = 0; dj < 4; ++dj) {
      z[dj] = wmma_f16(pa0, load_b(Vq + (dj * 16) * 64, 64, 0), z[dj]);
      z[dj] = wmma_f16(pa1, load_b(Vq + (dj * 16) * 64, 64, 32), z[dj]);
    }
    __syncthreads();   // reads of this block's tiles & P complete
  }

  // ---- normalize and scatter into concat-heads layout [B,S,H*D] ----
#pragma unroll
  for (int dj = 0; dj < 4; ++dj)
#pragma unroll
    for (int r = 0; r < 8; ++r) {
      const int m = r + 8 * g;
      const float zv = z[dj][r] / lrun[r];
      Zc[((size_t)b * S_ + s0 + m) * (H_ * D_) + h * D_ + dj * 16 + n] =
          (_Float16)zv;
    }
}

// ------------------------- output projection -------------------------------
// M=B*S=8192, N=E=1024, K=HD=1024, 16x64 tile per wave, double-buffered.
__global__ void k_outproj(const _Float16* __restrict__ Zc,
                          const _Float16* __restrict__ Wot,
                          const float* __restrict__ bo,
                          float* __restrict__ out) {
  const int lane = threadIdx.x & 31;
  const int wave = threadIdx.x >> 5;
  const int idx = blockIdx.x * 4 + wave;   // 0..8191
  const int mt = idx >> 4;
  const int ng = idx & 15;
  const int row0 = mt * 16;
  const int e0 = ng * 64;
  const int n = lane & 15, g = lane >> 4;

  const _Float16* arow = Zc + (size_t)row0 * (H_ * D_);
  v8f c[4] = {};
  v16h a = load_a(arow, H_ * D_, 0);
  v16h bf[4];
#pragma unroll
  for (int nj = 0; nj < 4; ++nj)
    bf[nj] = load_b(Wot + (size_t)(e0 + nj * 16) * (H_ * D_), H_ * D_, 0);

  for (int k0 = 0; k0 < H_ * D_; k0 += 32) {
    const int kn = k0 + 32;
    v16h a2;
    v16h b2[4];
    if (kn < H_ * D_) {
      a2 = load_a(arow, H_ * D_, kn);
#pragma unroll
      for (int nj = 0; nj < 4; ++nj)
        b2[nj] = load_b(Wot + (size_t)(e0 + nj * 16) * (H_ * D_), H_ * D_, kn);
    }
#pragma unroll
    for (int nj = 0; nj < 4; ++nj) c[nj] = wmma_f16(a, bf[nj], c[nj]);
    if (kn < H_ * D_) {
      a = a2;
#pragma unroll
      for (int nj = 0; nj < 4; ++nj) bf[nj] = b2[nj];
    }
  }

#pragma unroll
  for (int nj = 0; nj < 4; ++nj) {
    const float bv = bo[e0 + nj * 16 + n];
#pragma unroll
    for (int r = 0; r < 8; ++r) {
      const int m = r + 8 * g;
      out[(size_t)(row0 + m) * E_ + e0 + nj * 16 + n] = c[nj][r] + bv;
    }
  }
}

// ------------------------- launcher ----------------------------------------

extern "C" void kernel_launch(void* const* d_in, const int* in_sizes, int n_in,
                              void* d_out, int out_size, void* d_ws, size_t ws_size,
                              hipStream_t stream) {
  const float* x  = (const float*)d_in[0];
  const float* Wq = (const float*)d_in[1];
  const float* Wk = (const float*)d_in[2];
  const float* Wv = (const float*)d_in[3];
  const float* bq = (const float*)d_in[4];
  const float* bk = (const float*)d_in[5];
  const float* bv = (const float*)d_in[6];
  const float* Wo = (const float*)d_in[7];
  const float* bo = (const float*)d_in[8];
  float* out = (float*)d_out;

  const size_t NX   = (size_t)B_ * S_ * E_;        // 8,388,608
  const size_t NW   = (size_t)H_ * E_ * D_;        // 1,048,576
  const size_t NWO  = (size_t)(H_ * D_) * E_;      // 1,048,576
  const size_t NQKV = (size_t)B_ * H_ * S_ * D_;   // 8,388,608

  char* ws = (char*)d_ws;
  _Float16* Xh  = (_Float16*)ws;            ws += NX  * 2;
  _Float16* Wqt = (_Float16*)ws;            ws += NW  * 2;
  _Float16* Wkt = (_Float16*)ws;            ws += NW  * 2;
  _Float16* Wvt = (_Float16*)ws;            ws += NW  * 2;
  _Float16* Wot = (_Float16*)ws;            ws += NWO * 2;
  _Float16* Qh  = (_Float16*)ws;            ws += NQKV * 2;
  _Float16* Kh  = (_Float16*)ws;            ws += NQKV * 2;
  _Float16* Vt  = (_Float16*)ws;            ws += NQKV * 2;
  _Float16* Zc  = (_Float16*)ws;            ws += NX * 2;

  // 1) precision conversion + operand-layout transposes
  k_cvt_f16<<<(int)((NX + 255) / 256), 256, 0, stream>>>(x, Xh, (int)NX);
  k_tr_w  <<<(int)(NW  / 256), 256, 0, stream>>>(Wq, Wqt);
  k_tr_w  <<<(int)(NW  / 256), 256, 0, stream>>>(Wk, Wkt);
  k_tr_w  <<<(int)(NW  / 256), 256, 0, stream>>>(Wv, Wvt);
  k_tr_wo <<<(int)(NWO / 256), 256, 0, stream>>>(Wo, Wot);

  // 2) QKV projections (WMMA GEMM, Q/K/V split across grid.y)
  dim3 qkvGrid(B_ * H_ * (S_ / 64), 3);
  k_qkv<<<qkvGrid, 128, 0, stream>>>(Xh, Wqt, Wkt, Wvt, bq, bk, bv,
                                     Qh, Kh, Vt);

  // 3) flash attention (async K/V staging + WMMA QK^T / PV)
  k_attn<<<B_ * H_ * (S_ / 64), 128, 0, stream>>>(Qh, Kh, Vt, Zc);

  // 4) output projection (WMMA GEMM) + bias
  k_outproj<<<(B_ * S_ / 16) * (E_ / 64) / 4, 128, 0, stream>>>(Zc, Wot, bo, out);
}
